// SlotAttention_43327630082339
// MI455X (gfx1250) — matrix operand order repaired and checked by reading, not compiled, not run on hardware
//
#include <hip/hip_runtime.h>
#include <hip/hip_bf16.h>

typedef __attribute__((ext_vector_type(16))) __bf16 v16bf;
typedef __attribute__((ext_vector_type(8)))  float  v8f;

#define B_   32
#define NKV  4096
#define NQ   8
#define D_   256
#define H_   4
#define HD   64
#define MLPH 1024
#define EPS_ 1e-8f

// ---- WMMA fragment position helpers (CDNA5 ISA 7.12.2, bf16 16x16x32) ----
// A (16x32, MxK): lane<16: row=lane, K in {0..7,16..23}; lane>=16: row=lane-16, K in {8..15,24..31}
__device__ __forceinline__ void a_pos(int m, int k, int& lane, int& e) {
  int hi8 = (k >> 3) & 1;
  lane = m + 16 * hi8;
  int kb = 8 * hi8;
  e = (k < 16) ? (k - kb) : (8 + (k - 16 - kb));
}
// B (32x16, KxN): lane<16: col=lane, K in {0..7,16..23}; lane>=16: col=lane-16, K in {8..15,24..31}
__device__ __forceinline__ void b_pos(int k, int n, int& lane, int& e) {
  int hi8 = (k >> 3) & 1;
  lane = n + 16 * hi8;
  int kb = 8 * hi8;
  e = (k < 16) ? (k - kb) : (8 + (k - 16 - kb));
}

__device__ __forceinline__ v8f wmma_bf16(v16bf a, v16bf b, v8f c) {
  return __builtin_amdgcn_wmma_f32_16x16x32_bf16(false, a, false, b, (short)0, c,
                                                 false, false);
}

// ===== Kernel 0: one-shot Wk/Wv -> bf16 B-fragment layout =====
// Wf[ntile(32)][kk(8)][lane(32)][e(16)], n: 0..255 = K-proj, 256..511 = V-proj
__global__ __launch_bounds__(256) void k_prep_w(
    const float* __restrict__ Wk, const float* __restrict__ Wv, __bf16* __restrict__ Wf)
{
  int idx = blockIdx.x * 256 + threadIdx.x;     // 512*256 elements
  int n = idx >> 8, k = idx & 255;
  float w = (n < D_) ? Wk[(size_t)n * D_ + k] : Wv[(size_t)(n - D_) * D_ + k];
  int ntile = n >> 4, nn = n & 15, kk = k >> 5, kin = k & 31;
  int lane, e; b_pos(kin, nn, lane, e);
  Wf[(((size_t)ntile * 8 + kk) * 32 + lane) * 16 + e] = (__bf16)w;
}

// ===== Kernel 1: LN(inputs) + K/V projection (bf16 WMMA), fragment-layout out =====
// grid = B*NKV/16, 256 threads (8 waves). Block: 16 rows x 512 cols.
__global__ __launch_bounds__(256) void k_ln_kvproj(
    const float* __restrict__ x, const float* __restrict__ g, const float* __restrict__ be,
    const __bf16* __restrict__ Wf, __bf16* __restrict__ Kbf, __bf16* __restrict__ Vbf)
{
  __shared__ v16bf xsf[8][32];                    // A-fragment-ready LN(x): [kk][lane] 32B
  __shared__ float redA[16][16], redB[16][16];
  __shared__ float mean_[16], rstd_[16];
  const int t = threadIdx.x;
  const int row0 = blockIdx.x * 16;
  const int r = t >> 4, cg = t & 15;

  float vals[16];
  float s = 0.f, s2 = 0.f;
  const float* xr = x + (size_t)(row0 + r) * D_;
  #pragma unroll
  for (int i = 0; i < 16; ++i) {
    float v = xr[cg * 16 + i];
    vals[i] = v; s += v; s2 += v * v;
  }
  redA[r][cg] = s; redB[r][cg] = s2;
  __syncthreads();
  if (t < 16) {
    float m = 0.f, q = 0.f;
    for (int i = 0; i < 16; ++i) { m += redA[t][i]; q += redB[t][i]; }
    m *= (1.f / D_);
    float var = q * (1.f / D_) - m * m;
    mean_[t] = m; rstd_[t] = rsqrtf(var + 1e-5f);
  }
  __syncthreads();
  {
    float m = mean_[r], rs = rstd_[r];
    #pragma unroll
    for (int i = 0; i < 16; ++i) {
      int col = cg * 16 + i;
      float v = ((vals[i] - m) * rs) * g[col] + be[col];
      int kk = col >> 5, kin = col & 31;
      int lane, e; a_pos(r, kin, lane, e);
      ((__bf16*)&xsf[kk][lane])[e] = (__bf16)v;
    }
  }
  __syncthreads();

  const int wave = t >> 5, lane = t & 31;
  const int row = lane & 15;
  const int mbase = (lane < 16) ? 0 : 8;
  const v16bf* Wfv = (const v16bf*)Wf;
  v8f z = {0,0,0,0,0,0,0,0};
  v8f acc[4];
  #pragma unroll
  for (int nt = 0; nt < 4; ++nt) acc[nt] = z;

  for (int kk = 0; kk < 8; ++kk) {
    v16bf a = xsf[kk][lane];
    #pragma unroll
    for (int nt = 0; nt < 4; ++nt) {
      int ntile = wave * 4 + nt;
      v16bf bf = Wfv[((size_t)ntile * 8 + kk) * 32 + lane];
      acc[nt] = wmma_bf16(a, bf, acc[nt]);
    }
  }
  // Scatter D tiles into consumer fragment layouts.
  #pragma unroll
  for (int nt = 0; nt < 4; ++nt) {
    int n = wave * 64 + nt * 16 + row;            // output feature (0..511)
    #pragma unroll
    for (int j = 0; j < 8; ++j) {
      int gr = row0 + mbase + j;                  // global token row
      if (n < D_) {
        // K: A-fragment layout for k_attn: [ttile][h][kk2][lane][e]
        int h = n >> 6, kk2 = (n >> 5) & 1, kin = n & 31;
        int la, e; a_pos(gr & 15, kin, la, e);
        size_t ttile = (size_t)(gr >> 4);
        Kbf[(((ttile * H_ + h) * 2 + kk2) * 32 + la) * 16 + e] =
            (__bf16)(acc[nt][j] * 0.125f);        // hd^-0.5
      } else {
        // V: B-fragment layout for k_updates: [gchunk][h][nt4][lane][e]
        int nv = n - D_;
        int h = nv >> 6, nt4 = (nv >> 4) & 3, nn = nv & 15;
        int kin2 = gr & 31;
        int la, e; b_pos(kin2, nn, la, e);
        size_t gchunk = (size_t)(gr >> 5);
        Vbf[(((gchunk * H_ + h) * 4 + nt4) * 32 + la) * 16 + e] = (__bf16)acc[nt][j];
      }
    }
  }
}

// ===== Kernel 2: LN(slots) + q projection into zero-padded B-fragment layout =====
// qbf[b][h][kk(2)][lane(32)][e(16)]; lanes for slots 8..15 stay zero.
__global__ __launch_bounds__(256) void k_slot_q(
    const float* __restrict__ slots, const float* __restrict__ g, const float* __restrict__ be,
    const float* __restrict__ Wq, __bf16* __restrict__ qbf, float* __restrict__ colsum)
{
  __shared__ float sln[NQ][D_];
  __shared__ float redA[NQ][32], redB[NQ][32];
  __shared__ float mean_[NQ], rstd_[NQ];
  const int b = blockIdx.x;
  const int t = threadIdx.x;
  const int r = t >> 5, ln = t & 31;
  const float* sr = slots + ((size_t)b * NQ + r) * D_;
  float vals[8];
  float s = 0.f, s2 = 0.f;
  #pragma unroll
  for (int i = 0; i < 8; ++i) { float v = sr[ln * 8 + i]; vals[i] = v; s += v; s2 += v * v; }
  redA[r][ln] = s; redB[r][ln] = s2;
  __syncthreads();
  if (t < NQ) {
    float m = 0.f, q = 0.f;
    for (int i = 0; i < 32; ++i) { m += redA[t][i]; q += redB[t][i]; }
    m *= (1.f / D_);
    float var = q * (1.f / D_) - m * m;
    mean_[t] = m; rstd_[t] = rsqrtf(var + 1e-5f);
  }
  __syncthreads();
  {
    float m = mean_[r], rs = rstd_[r];
    #pragma unroll
    for (int i = 0; i < 8; ++i) {
      int c = ln * 8 + i;
      sln[r][c] = (vals[i] - m) * rs * g[c] + be[c];
    }
  }
  if (t < 32) colsum[b * 32 + t] = 0.f;
  // zero-fill this batch's q fragment region (pads slots 8..15 with 0)
  __bf16* qb_b = qbf + (size_t)b * (H_ * 2 * 32 * 16);
  for (int o = t; o < H_ * 2 * 32 * 16; o += 256) qb_b[o] = (__bf16)0.f;
  __syncthreads();
  for (int o = t; o < NQ * D_; o += 256) {
    int n = o >> 8;
    int c = o & 255;
    const float* w = Wq + (size_t)c * D_;
    float acc = 0.f;
    for (int k = 0; k < D_; ++k) acc += sln[n][k] * w[k];
    int h = c >> 6, kk = (c >> 5) & 1, kin = c & 31;
    int la, e; b_pos(kin, n, la, e);
    qb_b[((h * 2 + kk) * 32 + la) * 16 + e] = (__bf16)acc;
  }
}

// ===== Kernel 3: logits (WMMA) + joint softmax + colsum; attn stored transposed =====
// grid = B*16 blocks (256 tokens each), 256 threads. attn_t[b][hq(32)][token(4096)]
__global__ __launch_bounds__(256) void k_attn(
    const __bf16* __restrict__ Kbf, const __bf16* __restrict__ qbf,
    float* __restrict__ attn_t, float* __restrict__ colsum,
    float* __restrict__ avis, float* __restrict__ atemp, int write_vis)
{
  __shared__ float lg[256][32];
  const int b    = blockIdx.x >> 4;
  const int tok0 = (blockIdx.x & 15) * 256;
  const int t = threadIdx.x;
  const int wave = t >> 5, lane = t & 31;
  const int row = lane & 15, mbase = (lane < 16) ? 0 : 8;
  const v16bf* Kv = (const v16bf*)Kbf;
  const v16bf* Qv = (const v16bf*)qbf;
  const size_t ttile0 = (size_t)(b * NKV + tok0) >> 4;

  for (int mt2 = 0; mt2 < 2; ++mt2) {
    int mt = wave * 2 + mt2;
    #pragma unroll
    for (int h = 0; h < H_; ++h) {
      v8f acc = {0,0,0,0,0,0,0,0};
      #pragma unroll
      for (int kk = 0; kk < 2; ++kk) {
        v16bf a  = Kv[(((ttile0 + mt) * H_ + h) * 2 + kk) * 32 + lane];
        v16bf bf = Qv[(((size_t)b * H_ + h) * 2 + kk) * 32 + lane];
        acc = wmma_bf16(a, bf, acc);
      }
      if (row < NQ) {
        #pragma unroll
        for (int j = 0; j < 8; ++j) lg[mt * 16 + mbase + j][h * 8 + row] = acc[j];
      }
    }
  }
  __syncthreads();
  // joint softmax over H_*NQ = 32 per token (one token per thread)
  {
    float mx = -1e30f;
    #pragma unroll
    for (int i = 0; i < 32; ++i) mx = fmaxf(mx, lg[t][i]);
    float ex[32];
    float sum = 0.f;
    #pragma unroll
    for (int i = 0; i < 32; ++i) { float e = __expf(lg[t][i] - mx); ex[i] = e; sum += e; }
    float inv = 1.f / sum;
    #pragma unroll
    for (int i = 0; i < 32; ++i) {
      float v = ex[i] * inv;
      lg[t][i] = v;
      attn_t[((size_t)(b * 32 + i)) * NKV + tok0 + t] = v;   // coalesced across t
    }
    if (write_vis) {
      #pragma unroll
      for (int qi = 0; qi < NQ; ++qi) {
        float v = 0.f;
        #pragma unroll
        for (int h = 0; h < H_; ++h) v += lg[t][h * 8 + qi];
        size_t oi = (size_t)(b * NKV + tok0 + t) * NQ + qi;
        avis[oi] = v; atemp[oi] = v;
      }
    }
  }
  __syncthreads();
  if (t < 32) {
    float s = 0.f;
    for (int m2 = 0; m2 < 256; ++m2) s += lg[m2][t];
    atomicAdd(&colsum[b * 32 + t], s + 256.f * EPS_);
  }
}

// ===== Kernel 4: updates = a_norm^T @ V (WMMA) =====
// grid = B*H blocks, 256 threads (8 waves splitting K=4096).
__global__ __launch_bounds__(256) void k_updates(
    const float* __restrict__ attn_t, const float* __restrict__ colsum,
    const __bf16* __restrict__ Vbf, float* __restrict__ upd)
{
  __shared__ float oa[4][16][16];
  const int b = blockIdx.x >> 2;
  const int h = blockIdx.x & 3;
  const int t = threadIdx.x;
  for (int i = t; i < 4 * 16 * 16; i += 256) ((float*)oa)[i] = 0.f;
  __syncthreads();
  const int wave = t >> 5, lane = t & 31;
  const int m = lane & 15;
  const int hi8 = (lane < 16) ? 0 : 1;
  const int kb = 8 * hi8;
  const int mbase = (lane < 16) ? 0 : 8;
  const v16bf* Vv = (const v16bf*)Vbf;

  float inv = 1.f;
  const float* arow = attn_t;
  if (m < NQ) {
    inv = 1.f / colsum[b * 32 + h * 8 + m];
    arow = attn_t + (size_t)(b * 32 + h * 8 + m) * NKV;
  }

  v8f z = {0,0,0,0,0,0,0,0};
  v8f acc[4];
  #pragma unroll
  for (int nt = 0; nt < 4; ++nt) acc[nt] = z;

  for (int kc = 0; kc < 16; ++kc) {
    int tok0 = wave * 512 + kc * 32;
    size_t gchunk = (size_t)b * 128 + wave * 16 + kc;
    if (kc + 1 < 16)
      __builtin_prefetch(&Vv[(((gchunk + 1) * H_ + h) * 4) * 32 + lane], 0, 1);
    v16bf a;
    if (m < NQ) {
      // A K-pattern = two runs of 8 consecutive tokens -> vectorizable loads
      #pragma unroll
      for (int e = 0; e < 8; ++e) {
        float av = arow[tok0 + kb + e];
        a[e] = (__bf16)((av + EPS_) * inv);
      }
      #pragma unroll
      for (int e = 0; e < 8; ++e) {
        float av = arow[tok0 + 16 + kb + e];
        a[8 + e] = (__bf16)((av + EPS_) * inv);
      }
    } else {
      #pragma unroll
      for (int e = 0; e < 16; ++e) a[e] = (__bf16)0.f;
    }
    #pragma unroll
    for (int nt = 0; nt < 4; ++nt) {
      v16bf bf = Vv[((gchunk * H_ + h) * 4 + nt) * 32 + lane];
      acc[nt] = wmma_bf16(a, bf, acc[nt]);
    }
  }
  #pragma unroll
  for (int nt = 0; nt < 4; ++nt)
    #pragma unroll
    for (int j = 0; j < 8; ++j)
      atomicAdd(&oa[nt][mbase + j][lane & 15], acc[nt][j]);
  __syncthreads();
  for (int o = t; o < NQ * HD; o += 256) {
    int qi = o >> 6, d = o & 63;
    upd[(size_t)(b * NQ + qi) * D_ + h * HD + d] = oa[d >> 4][qi][d & 15];
  }
}

// ===== Kernel 5: GRU cell + LN + MLP residual (256 rows, scalar f32) =====
__global__ __launch_bounds__(256) void k_gru_mlp(
    const float* __restrict__ slots_prev, const float* __restrict__ upd,
    const float* __restrict__ W_ih, const float* __restrict__ W_hh,
    const float* __restrict__ b_ih, const float* __restrict__ b_hh,
    const float* __restrict__ lgm, const float* __restrict__ lbm,
    const float* __restrict__ W1, const float* __restrict__ b1,
    const float* __restrict__ W2, const float* __restrict__ b2,
    float* __restrict__ slots_out, float* __restrict__ out_final, int last)
{
  __shared__ float sp[NQ][D_];
  __shared__ float up[NQ][D_];
  __shared__ float sn[NQ][D_];
  __shared__ float hid[NQ][MLPH];
  __shared__ float redA[NQ][32], redB[NQ][32];
  __shared__ float mean_[NQ], rstd_[NQ];
  const int b = blockIdx.x;
  const int t = threadIdx.x;
  for (int o = t; o < NQ * D_; o += 256) {
    sp[o >> 8][o & 255] = slots_prev[(size_t)b * NQ * D_ + o];
    up[o >> 8][o & 255] = upd[(size_t)b * NQ * D_ + o];
  }
  __syncthreads();
  for (int o = t; o < NQ * D_; o += 256) {
    int n = o >> 8, c = o & 255;
    const float* wir = W_ih + (size_t)c * D_;
    const float* wiz = W_ih + (size_t)(256 + c) * D_;
    const float* win = W_ih + (size_t)(512 + c) * D_;
    const float* whr = W_hh + (size_t)c * D_;
    const float* whz = W_hh + (size_t)(256 + c) * D_;
    const float* whn = W_hh + (size_t)(512 + c) * D_;
    float xr = b_ih[c], xz = b_ih[256 + c], xn = b_ih[512 + c];
    float hr = b_hh[c], hz = b_hh[256 + c], hn = b_hh[512 + c];
    for (int k = 0; k < D_; ++k) {
      float u = up[n][k], s = sp[n][k];
      xr += u * wir[k]; xz += u * wiz[k]; xn += u * win[k];
      hr += s * whr[k]; hz += s * whz[k]; hn += s * whn[k];
    }
    float r = 1.f / (1.f + __expf(-(xr + hr)));
    float zz = 1.f / (1.f + __expf(-(xz + hz)));
    float nn = tanhf(xn + r * hn);
    sn[n][c] = (1.f - zz) * nn + zz * sp[n][c];
  }
  __syncthreads();
  {
    int r = t >> 5, ln = t & 31;
    float s = 0.f, s2 = 0.f;
    #pragma unroll
    for (int i = 0; i < 8; ++i) { float v = sn[r][ln * 8 + i]; s += v; s2 += v * v; }
    redA[r][ln] = s; redB[r][ln] = s2;
  }
  __syncthreads();
  if (t < NQ) {
    float m = 0.f, q = 0.f;
    for (int i = 0; i < 32; ++i) { m += redA[t][i]; q += redB[t][i]; }
    m *= (1.f / D_);
    float var = q * (1.f / D_) - m * m;
    mean_[t] = m; rstd_[t] = rsqrtf(var + 1e-5f);
  }
  __syncthreads();
  {
    int r = t >> 5, ln = t & 31;
    float m = mean_[r], rs = rstd_[r];
    #pragma unroll
    for (int i = 0; i < 8; ++i) {
      int c = ln * 8 + i;
      up[r][c] = (sn[r][c] - m) * rs * lgm[c] + lbm[c];   // reuse `up` as LN(sn)
    }
  }
  __syncthreads();
  for (int o = t; o < NQ * MLPH; o += 256) {
    int n = o >> 10, j = o & 1023;
    const float* w = W1 + (size_t)j * D_;
    float acc = b1[j];
    for (int k = 0; k < D_; ++k) acc += up[n][k] * w[k];
    hid[n][j] = fmaxf(acc, 0.f);
  }
  __syncthreads();
  for (int o = t; o < NQ * D_; o += 256) {
    int n = o >> 8, c = o & 255;
    const float* w = W2 + (size_t)c * MLPH;
    float acc = b2[c];
    for (int k = 0; k < MLPH; ++k) acc += hid[n][k] * w[k];
    float v = sn[n][c] + acc;
    slots_out[(size_t)b * NQ * D_ + o] = v;
    if (last) out_final[(size_t)b * NQ * D_ + o] = v;
  }
}

// ===================== host launcher =====================
extern "C" void kernel_launch(void* const* d_in, const int* in_sizes, int n_in,
                              void* d_out, int out_size, void* d_ws, size_t ws_size,
                              hipStream_t stream)
{
  (void)in_sizes; (void)n_in; (void)out_size; (void)ws_size;
  const float* inputs   = (const float*)d_in[0];
  const float* slots_in = (const float*)d_in[1];
  const float* ln_in_g  = (const float*)d_in[2];
  const float* ln_in_b  = (const float*)d_in[3];
  const float* ln_sl_g  = (const float*)d_in[4];
  const float* ln_sl_b  = (const float*)d_in[5];
  const float* ln_ml_g  = (const float*)d_in[6];
  const float* ln_ml_b  = (const float*)d_in[7];
  const float* Wq   = (const float*)d_in[8];
  const float* Wk   = (const float*)d_in[9];
  const float* Wv   = (const float*)d_in[10];
  const float* W_ih = (const float*)d_in[11];
  const float* W_hh = (const float*)d_in[12];
  const float* b_ih = (const float*)d_in[13];
  const float* b_hh = (const float*)d_in[14];
  const float* W1   = (const float*)d_in[15];
  const float* b1   = (const float*)d_in[16];
  const float* W2   = (const float*)d_in[17];
  const float* b2   = (const float*)d_in[18];

  char* ws = (char*)d_ws;
  size_t off = 0;
  __bf16* Kbf = (__bf16*)(ws + off);  off += (size_t)B_ * NKV * D_ * 2;   // A-frag layout
  __bf16* Vbf = (__bf16*)(ws + off);  off += (size_t)B_ * NKV * D_ * 2;   // B-frag layout
  float* attn_t = (float*)(ws + off); off += (size_t)B_ * 32 * NKV * 4;   // [b][hq][tok]
  __bf16* qbf = (__bf16*)(ws + off);  off += (size_t)B_ * H_ * 2 * 32 * 16 * 2;
  __bf16* Wf  = (__bf16*)(ws + off);  off += (size_t)512 * 256 * 2;
  float* colsum = (float*)(ws + off); off += (size_t)B_ * 32 * 4;
  float* upd = (float*)(ws + off);    off += (size_t)B_ * NQ * D_ * 4;
  float* slots_ws = (float*)(ws + off); off += (size_t)B_ * NQ * D_ * 4;

  float* out_slots = (float*)d_out;
  float* avis  = out_slots + (size_t)B_ * NQ * D_;
  float* atemp = avis + (size_t)B_ * NKV * NQ;

  k_prep_w<<<512, 256, 0, stream>>>(Wk, Wv, Wf);
  k_ln_kvproj<<<B_ * NKV / 16, 256, 0, stream>>>(inputs, ln_in_g, ln_in_b, Wf, Kbf, Vbf);

  for (int it = 0; it < 3; ++it) {
    const float* sp = (it == 0) ? slots_in : slots_ws;
    int last = (it == 2) ? 1 : 0;
    k_slot_q<<<B_, 256, 0, stream>>>(sp, ln_sl_g, ln_sl_b, Wq, qbf, colsum);
    k_attn<<<B_ * 16, 256, 0, stream>>>(Kbf, qbf, attn_t, colsum, avis, atemp, last);
    k_updates<<<B_ * H_, 256, 0, stream>>>(attn_t, colsum, Vbf, upd);
    k_gru_mlp<<<B_, 256, 0, stream>>>(sp, upd, W_ih, W_hh, b_ih, b_hh,
                                      ln_ml_g, ln_ml_b, W1, b1, W2, b2,
                                      slots_ws, out_slots, last);
  }
}
